// SwinTransformerBlock_52183852646686
// MI455X (gfx1250) — matrix operand order repaired and checked by reading, not compiled, not run on hardware
//
#include <hip/hip_runtime.h>
#include <hip/hip_bf16.h>

// ---------------------------------------------------------------------------
// Swin Transformer block for MI455X (gfx1250), wave32 + WMMA bf16.
// B=32, R=56, S=8, C=128, H=4, HD=32, N=64, NW=7, MLP=512
//
// B-operand matrices are stored FRAGMENT-MAJOR: for each 32(K)x16(N) tile,
// lane L's 16 bf16 values (k = kb..kb+15 at column n, kb = L<16?0:16,
// n = L&15) are 32 contiguous bytes at tile*1024 + L*32.
//   -> B fragment = 2x b128 loads (global or ds), zero packing ALU.
// A-operand matrices are row-major bf16 -> 2x ds_load_b128 per fragment.
// Per-wave work is assigned so A fragments are loop-invariant and hoisted.
// Relative-position bias + shift mask are pre-baked into a per-class table
// and injected through the WMMA C operand (free accumulate).
// ---------------------------------------------------------------------------

typedef __attribute__((ext_vector_type(16))) __bf16        v16bf;
typedef __attribute__((ext_vector_type(8)))  float         v8f;
typedef __attribute__((ext_vector_type(8)))  unsigned int  v8u;

__device__ static inline v8f wmma_bf16(v16bf a, v16bf b, v8f c) {
    return __builtin_amdgcn_wmma_f32_16x16x32_bf16(false, a, false, b,
                                                   (short)0, c, false, false);
}

// A fragment (16x32 bf16) from row-major storage, leading dim ld.
__device__ static inline v16bf load_a16(const __bf16* src, int ld) {
    int lane = threadIdx.x & 31;
    int m = lane & 15;
    int kb = (lane < 16) ? 0 : 16;
    const __bf16* r = src + m * ld;
    v16bf a;
#pragma unroll
    for (int v = 0; v < 8; ++v) {
        int k = kb + 2 * v;
        a[2 * v]     = r[k];
        a[2 * v + 1] = r[k + 1];
    }
    return a;
}

// B fragment from fragment-major storage: tile base in uint units, 256/tile.
__device__ static inline v16bf load_b_fragG(const unsigned int* tile_base) {
    int lane = threadIdx.x & 31;
    const v8u* p = (const v8u*)tile_base;
    return __builtin_bit_cast(v16bf, p[lane]);
}
__device__ static inline v16bf load_b_fragL(const __bf16* tile_base) {
    int lane = threadIdx.x & 31;
    const v8u* p = (const v8u*)tile_base;
    return __builtin_bit_cast(v16bf, p[lane]);
}

__device__ static inline unsigned int packbf(float lo, float hi) {
    unsigned short l = __builtin_bit_cast(unsigned short, (__bf16)lo);
    unsigned short h = __builtin_bit_cast(unsigned short, (__bf16)hi);
    return (unsigned int)l | ((unsigned int)h << 16);
}

// fragment-major packing index helper: region-local uint index o ->
// (k, n) for a KxN weight with NT = N/16 tiles per k-chunk (NT power of 2).
__device__ static inline void frag_kn(int o, int ntshift, int& k, int& n) {
    int tile = o >> 8, r = o & 255, L = r >> 3, vv = r & 7;
    int ks = tile >> ntshift, nt = tile & ((1 << ntshift) - 1);
    k = ks * 32 + ((L >> 4) << 4) + 2 * vv;
    n = nt * 16 + (L & 15);
}

// ---------------------------------------------------------------------------
// Kernel 1: weight prep (fragment-major bf16 pairs), fold wo@wp, bake
// combined bias+mask table, fold q-scale.
// ws uint32: wq*s[8192] wk[8192] wv[8192] wproj[8192] w1[32768] w2[32768]
// then f32: cbias[4cls][4h][64][64] @393216B, bproj[128] @655360B,
//           bqs[128] @655872B
// ---------------------------------------------------------------------------
__global__ void prep_kernel(const float* __restrict__ wq, const float* __restrict__ wk,
                            const float* __restrict__ wv, const float* __restrict__ wo,
                            const float* __restrict__ wp, const float* __restrict__ bo,
                            const float* __restrict__ bp, const float* __restrict__ w1,
                            const float* __restrict__ w2, const float* __restrict__ bias_table,
                            const int* __restrict__ rel_pos, const float* __restrict__ bq,
                            const unsigned char* __restrict__ attn_mask,
                            unsigned int* __restrict__ wsp, float* __restrict__ cbias,
                            float* __restrict__ bproj, float* __restrict__ bqs) {
    const float qscale = 0.17677669529663688f;  // 1/sqrt(32)
    const int total = 164096;
    for (int i = blockIdx.x * blockDim.x + threadIdx.x; i < total;
         i += gridDim.x * blockDim.x) {
        if (i < 8192) {                       // wq * qscale, frag-major
            int k, n; frag_kn(i, 3, k, n);
            wsp[i] = packbf(wq[k * 128 + n] * qscale, wq[(k + 1) * 128 + n] * qscale);
        } else if (i < 16384) {               // wk
            int k, n; frag_kn(i - 8192, 3, k, n);
            wsp[i] = packbf(wk[k * 128 + n], wk[(k + 1) * 128 + n]);
        } else if (i < 24576) {               // wv
            int k, n; frag_kn(i - 16384, 3, k, n);
            wsp[i] = packbf(wv[k * 128 + n], wv[(k + 1) * 128 + n]);
        } else if (i < 32768) {               // wproj = wo_flat @ wp
            int k, n; frag_kn(i - 24576, 3, k, n);
            float a0 = 0.f, a1 = 0.f;
            for (int t = 0; t < 128; ++t) {
                float wpc = wp[t * 128 + n];
                a0 += wo[k * 128 + t] * wpc;
                a1 += wo[(k + 1) * 128 + t] * wpc;
            }
            wsp[i] = packbf(a0, a1);
        } else if (i < 65536) {               // w1: 128x512 (NT=32)
            int k, n; frag_kn(i - 32768, 5, k, n);
            wsp[i] = packbf(w1[k * 512 + n], w1[(k + 1) * 512 + n]);
        } else if (i < 98304) {               // w2: 512x128 (NT=8)
            int k, n; frag_kn(i - 65536, 3, k, n);
            wsp[i] = packbf(w2[k * 128 + n], w2[(k + 1) * 128 + n]);
        } else if (i < 163840) {              // cbias[cls][h][q][k]
            int o = i - 98304;
            int cls = o >> 14, rem = o & 16383;
            int h = rem >> 12, qk = rem & 4095;
            int r0 = rel_pos[qk], r1 = rel_pos[4096 + qk];
            float bias = bias_table[h * 225 + r0 * 15 + r1];
            int wi = (cls & 2) ? 6 : 0, wj = (cls & 1) ? 6 : 0;
            unsigned char m = attn_mask[(size_t)(wi * 7 + wj) * 4096 + qk];
            cbias[o] = m ? bias : -1e30f;
        } else if (i < 163968) {              // bproj = bo @ wp + bp
            int j = i - 163840;
            float acc = bp[j];
            for (int c = 0; c < 128; ++c) acc += bo[c] * wp[c * 128 + j];
            bproj[j] = acc;
        } else {                              // bqs = bq * qscale
            int j = i - 163968;
            bqs[j] = bq[j] * qscale;
        }
    }
}

// ---------------------------------------------------------------------------
// Kernel 2: LN1 + shift + windowed attention + merged projection + residual.
// One block (8 wave32) per window. 64KB LDS, aliased:
//   [0      ,  8192) X  (64x128 bf16, A-layout)  -> reused as P heads 0,1
//   [8192   , 16384) Q  (64x128 bf16, A-layout)  -> reused as P heads 2,3
//   [16384  , 24576) KT frag-major: tile(h*4+ttok) x 32L x 16e -> reused as O
//   [24576  , 32768) V  frag-major: tile(ks*8+ct) x 32L x 16e
// ---------------------------------------------------------------------------
__global__ __launch_bounds__(256) void attn_kernel(
    const float* __restrict__ emb, const float* __restrict__ ln1s,
    const float* __restrict__ ln1b, const float* __restrict__ bqs,
    const float* __restrict__ bk, const float* __restrict__ bv,
    const unsigned int* __restrict__ wqP, const unsigned int* __restrict__ wkP,
    const unsigned int* __restrict__ wvP, const unsigned int* __restrict__ wprojP,
    const float* __restrict__ cbias, const float* __restrict__ bproj,
    float* __restrict__ out) {
    __shared__ __bf16 smem[32768];
    __bf16* X   = smem;
    __bf16* Q   = smem + 8192;
    __bf16* KTh = smem + 16384;   // frag-major K^T (bf16 element view)
    __bf16* Vh  = smem + 24576;   // frag-major V   (bf16 element view)
    __bf16* P   = smem;           // aliases X+Q (4 heads x 64 x 64, A-layout)
    __bf16* O   = smem + 16384;   // aliases KT  (64 x 128, A-layout)

    const int blk = blockIdx.x;
    const int wjw = blk % 7;
    const int wiw = (blk / 7) % 7;
    const int b   = blk / 49;
    const int lane = threadIdx.x & 31;
    const int wv8  = threadIdx.x >> 5;

    __builtin_prefetch(wqP, 0, 3);
    __builtin_prefetch(wkP, 0, 3);
    __builtin_prefetch(wvP, 0, 3);
    __builtin_prefetch(wprojP, 0, 3);

    // ------------- Phase 1: gather (shifted) rows + LayerNorm1 -> X
    {
        float s0 = ln1s[lane], s1 = ln1s[lane + 32], s2 = ln1s[lane + 64], s3 = ln1s[lane + 96];
        float c0 = ln1b[lane], c1 = ln1b[lane + 32], c2 = ln1b[lane + 64], c3 = ln1b[lane + 96];
        for (int rr = 0; rr < 8; ++rr) {
            int row = wv8 * 8 + rr;
            int qv = row >> 3, r = row & 7;
            int oi = wiw * 8 + qv + 52; if (oi >= 56) oi -= 56;
            int oj = wjw * 8 + r  + 52; if (oj >= 56) oj -= 56;
            const float* src = emb + (((size_t)b * 56 + oi) * 56 + oj) * 128;
            float x0 = src[lane], x1 = src[lane + 32], x2 = src[lane + 64], x3 = src[lane + 96];
            float s  = x0 + x1 + x2 + x3;
            float sq = x0 * x0 + x1 * x1 + x2 * x2 + x3 * x3;
#pragma unroll
            for (int m = 1; m < 32; m <<= 1) {
                s  += __shfl_xor(s, m, 32);
                sq += __shfl_xor(sq, m, 32);
            }
            float mu   = s * (1.f / 128.f);
            float var  = sq * (1.f / 128.f) - mu * mu;
            float rstd = rsqrtf(var + 1e-6f);
            __bf16* dst = X + row * 128;
            dst[lane]      = (__bf16)((x0 - mu) * rstd * s0 + c0);
            dst[lane + 32] = (__bf16)((x1 - mu) * rstd * s1 + c1);
            dst[lane + 64] = (__bf16)((x2 - mu) * rstd * s2 + c2);
            dst[lane + 96] = (__bf16)((x3 - mu) * rstd * s3 + c3);
        }
    }
    __syncthreads();

    // ------------- Phase 2: Q,K,V = X @ W (+bias); A fragments hoisted.
    {
        const int n  = lane & 15;
        const int mb = (lane < 16) ? 0 : 8;
        const int mt  = wv8 >> 1;
        const int ntb = (wv8 & 1) * 4;
        v16bf ax[4];
#pragma unroll
        for (int ks = 0; ks < 4; ++ks)
            ax[ks] = load_a16(X + mt * 16 * 128 + ks * 32, 128);
#pragma unroll
        for (int mat = 0; mat < 3; ++mat) {
            const unsigned int* W = (mat == 0) ? wqP : (mat == 1) ? wkP : wvP;
            const float* bvec = (mat == 0) ? bqs : (mat == 1) ? bk : bv;
#pragma unroll
            for (int j = 0; j < 4; ++j) {
                int nt = ntb + j;
                v8f acc = {};
#pragma unroll
                for (int ks = 0; ks < 4; ++ks)
                    acc = wmma_bf16(ax[ks], load_b_fragG(W + (ks * 8 + nt) * 256), acc);
                int col = nt * 16 + n;
                float bias = bvec[col];
                if (mat == 0) {
#pragma unroll
                    for (int v = 0; v < 8; ++v)
                        Q[(mt * 16 + mb + v) * 128 + col] = (__bf16)(acc[v] + bias);
                } else if (mat == 1) {
                    // K^T frag-major: tile = h*4 + mt, L = (d>=16?16:0)+tok&15,
                    // elem = d&15; tok&15 = mb+v -> stride 16 elems per v.
                    int hh = col >> 5, d = col & 31;
                    __bf16* dst = KTh + ((hh * 4 + mt) * 32 + ((d >> 4) << 4) + mb) * 16
                                      + (d & 15);
#pragma unroll
                    for (int v = 0; v < 8; ++v)
                        dst[v * 16] = (__bf16)(acc[v] + bias);
                } else {
                    // V frag-major: tile = (mt>>1)*8 + col/16, L = (mt&1)*16 +
                    // col&15, elem = mb+v -> 8 contiguous bf16 (one b128 store)
                    __bf16* dst = Vh + (((mt >> 1) * 8 + (col >> 4)) * 32
                                        + ((mt & 1) << 4) + (col & 15)) * 16 + mb;
#pragma unroll
                    for (int v = 0; v < 8; ++v)
                        dst[v] = (__bf16)(acc[v] + bias);
                }
            }
        }
    }
    __syncthreads();

    // ------------- Phase 3: logits = Q Kt + cbias (bias injected via WMMA C)
    const int h  = wv8 >> 1;
    const int mh = wv8 & 1;
    const int n_  = lane & 15;
    const int mb_ = (lane < 16) ? 0 : 8;
    v8f lg[2][4];
    {
        int cls = ((wiw == 6) ? 2 : 0) + ((wjw == 6) ? 1 : 0);
        const float* cb = cbias + (size_t)(cls * 4 + h) * 4096;
#pragma unroll
        for (int mi = 0; mi < 2; ++mi) {
            v16bf a = load_a16(Q + (2 * mh + mi) * 16 * 128 + h * 32, 128);
#pragma unroll
            for (int nt = 0; nt < 4; ++nt) {
                // preload bias+mask in D-fragment layout and use as C operand
                v8f cinit;
#pragma unroll
                for (int v = 0; v < 8; ++v)
                    cinit[v] = cb[((2 * mh + mi) * 16 + mb_ + v) * 64 + nt * 16 + n_];
                lg[mi][nt] = wmma_bf16(a, load_b_fragL(KTh + (h * 4 + nt) * 512), cinit);
            }
        }
    }
    __syncthreads();  // all waves done reading Q/KT; P may overwrite them

    // ------------- Phase 4: in-register softmax, write P (bf16, A-layout)
    {
#pragma unroll
        for (int mi = 0; mi < 2; ++mi) {
#pragma unroll
            for (int v = 0; v < 8; ++v) {
                float rm = fmaxf(fmaxf(lg[mi][0][v], lg[mi][1][v]),
                                 fmaxf(lg[mi][2][v], lg[mi][3][v]));
#pragma unroll
                for (int m = 1; m < 16; m <<= 1) rm = fmaxf(rm, __shfl_xor(rm, m, 32));
                float rs = 0.f;
#pragma unroll
                for (int nt = 0; nt < 4; ++nt) {
                    float e = __expf(lg[mi][nt][v] - rm);
                    lg[mi][nt][v] = e;
                    rs += e;
                }
#pragma unroll
                for (int m = 1; m < 16; m <<= 1) rs += __shfl_xor(rs, m, 32);
                float inv = 1.f / rs;
                int grow = (2 * mh + mi) * 16 + mb_ + v;
#pragma unroll
                for (int nt = 0; nt < 4; ++nt)
                    P[h * 4096 + grow * 64 + nt * 16 + n_] =
                        (__bf16)(lg[mi][nt][v] * inv);
            }
        }
    }
    __syncthreads();

    // ------------- Phase 5: O = P @ V (K=64 tokens = 2 wmma steps)
    {
#pragma unroll
        for (int mi = 0; mi < 2; ++mi) {
            int mt = 2 * mh + mi;
            v16bf a0 = load_a16(P + h * 4096 + mt * 16 * 64, 64);
            v16bf a1 = load_a16(P + h * 4096 + mt * 16 * 64 + 32, 64);
#pragma unroll
            for (int nt = 0; nt < 2; ++nt) {
                int ct = h * 2 + nt;  // 16-col tile of V
                v8f acc = {};
                acc = wmma_bf16(a0, load_b_fragL(Vh + (0 * 8 + ct) * 512), acc);
                acc = wmma_bf16(a1, load_b_fragL(Vh + (1 * 8 + ct) * 512), acc);
                int col = ct * 16 + n_;
#pragma unroll
                for (int v = 0; v < 8; ++v)
                    O[(mt * 16 + mb_ + v) * 128 + col] = (__bf16)acc[v];
            }
        }
    }
    __syncthreads();

    // ------------- Phase 6: Y = O @ (wo@wp) + bproj; un-shift + residual
    {
        const int mt = wv8 >> 1, ng = wv8 & 1;
        v16bf ao[4];
#pragma unroll
        for (int ks = 0; ks < 4; ++ks)
            ao[ks] = load_a16(O + mt * 16 * 128 + ks * 32, 128);
        size_t rowaddr[8];
#pragma unroll
        for (int v = 0; v < 8; ++v) {
            int row = mt * 16 + mb_ + v;
            int qv = row >> 3, r = row & 7;
            int oi = wiw * 8 + qv + 52; if (oi >= 56) oi -= 56;
            int oj = wjw * 8 + r  + 52; if (oj >= 56) oj -= 56;
            rowaddr[v] = (((size_t)b * 56 + oi) * 56 + oj) * 128;
        }
#pragma unroll
        for (int j = 0; j < 4; ++j) {
            int nt = ng * 4 + j;
            v8f acc = {};
#pragma unroll
            for (int ks = 0; ks < 4; ++ks)
                acc = wmma_bf16(ao[ks], load_b_fragG(wprojP + (ks * 8 + nt) * 256), acc);
            int col = nt * 16 + n_;
            float bias = bproj[col];
#pragma unroll
            for (int v = 0; v < 8; ++v)
                out[rowaddr[v] + col] = emb[rowaddr[v] + col] + acc[v] + bias;
        }
    }
}

// ---------------------------------------------------------------------------
// Kernel 3: LN2 + MLP (128 -> 512 GELU -> 128), in-place on `io`.
// One block per 32 tokens. LDS: X2 8KB + H1 32KB = 40KB.
// ---------------------------------------------------------------------------
__global__ __launch_bounds__(256) void mlp_kernel(
    float* __restrict__ io, const float* __restrict__ ln2s,
    const float* __restrict__ ln2b, const unsigned int* __restrict__ w1P,
    const float* __restrict__ b1, const unsigned int* __restrict__ w2P,
    const float* __restrict__ b2) {
    __shared__ __bf16 X2[32 * 128];
    __shared__ __bf16 H1[32 * 512];
    const int t0 = blockIdx.x * 32;
    const int lane = threadIdx.x & 31;
    const int wv8  = threadIdx.x >> 5;

    __builtin_prefetch(w1P, 0, 3);
    __builtin_prefetch(w2P, 0, 3);

    // LayerNorm2: 4 rows per wave
    {
        float s0 = ln2s[lane], s1 = ln2s[lane + 32], s2 = ln2s[lane + 64], s3 = ln2s[lane + 96];
        float c0 = ln2b[lane], c1 = ln2b[lane + 32], c2 = ln2b[lane + 64], c3 = ln2b[lane + 96];
        for (int rr = 0; rr < 4; ++rr) {
            int row = wv8 * 4 + rr;
            const float* src = io + (size_t)(t0 + row) * 128;
            float x0 = src[lane], x1 = src[lane + 32], x2 = src[lane + 64], x3 = src[lane + 96];
            float s  = x0 + x1 + x2 + x3;
            float sq = x0 * x0 + x1 * x1 + x2 * x2 + x3 * x3;
#pragma unroll
            for (int m = 1; m < 32; m <<= 1) {
                s  += __shfl_xor(s, m, 32);
                sq += __shfl_xor(sq, m, 32);
            }
            float mu   = s * (1.f / 128.f);
            float var  = sq * (1.f / 128.f) - mu * mu;
            float rstd = rsqrtf(var + 1e-6f);
            __bf16* dst = X2 + row * 128;
            dst[lane]      = (__bf16)((x0 - mu) * rstd * s0 + c0);
            dst[lane + 32] = (__bf16)((x1 - mu) * rstd * s1 + c1);
            dst[lane + 64] = (__bf16)((x2 - mu) * rstd * s2 + c2);
            dst[lane + 96] = (__bf16)((x3 - mu) * rstd * s3 + c3);
        }
    }
    __syncthreads();

    const int n  = lane & 15;
    const int mb = (lane < 16) ? 0 : 8;

    // H1 = gelu(X2 @ W1 + b1): wave -> fixed mt, 8 consecutive ntiles
    {
        const int mt  = wv8 >> 2;
        const int ntb = (wv8 & 3) * 8;
        v16bf ax[4];
#pragma unroll
        for (int ks = 0; ks < 4; ++ks)
            ax[ks] = load_a16(X2 + mt * 16 * 128 + ks * 32, 128);
#pragma unroll
        for (int j = 0; j < 8; ++j) {
            int nt = ntb + j;
            v8f acc = {};
#pragma unroll
            for (int ks = 0; ks < 4; ++ks)
                acc = wmma_bf16(ax[ks], load_b_fragG(w1P + (ks * 32 + nt) * 256), acc);
            int col = nt * 16 + n;
            float bias = b1[col];
#pragma unroll
            for (int v = 0; v < 8; ++v) {
                float x = acc[v] + bias;
                // tanh-GELU: g = x * e/(e+1), e = exp(1.5957691*(x+0.044715x^3))
                float e = __expf(1.5957691216f * (x + 0.044715f * x * x * x));
                H1[(mt * 16 + mb + v) * 512 + col] = (__bf16)(x * (e / (e + 1.f)));
            }
        }
    }
    __syncthreads();

    // Y = H1 @ W2 + b2: wave -> fixed mt, 2 ntiles, A shared per k-step
    {
        const int mt  = wv8 >> 2;
        const int nt0 = (wv8 & 3) * 2, nt1 = nt0 + 1;
        v8f acc0 = {}, acc1 = {};
#pragma unroll
        for (int ks = 0; ks < 16; ++ks) {
            v16bf a = load_a16(H1 + mt * 16 * 512 + ks * 32, 512);
            acc0 = wmma_bf16(a, load_b_fragG(w2P + (ks * 8 + nt0) * 256), acc0);
            acc1 = wmma_bf16(a, load_b_fragG(w2P + (ks * 8 + nt1) * 256), acc1);
        }
        float bias0 = b2[nt0 * 16 + n];
        float bias1 = b2[nt1 * 16 + n];
#pragma unroll
        for (int v = 0; v < 8; ++v) {
            size_t rowoff = (size_t)(t0 + mt * 16 + mb + v) * 128;
            io[rowoff + nt0 * 16 + n] = acc0[v] + bias0;
            io[rowoff + nt1 * 16 + n] = acc1[v] + bias1;
        }
    }
}

// ---------------------------------------------------------------------------
extern "C" void kernel_launch(void* const* d_in, const int* in_sizes, int n_in,
                              void* d_out, int out_size, void* d_ws, size_t ws_size,
                              hipStream_t stream) {
    (void)in_sizes; (void)n_in; (void)out_size; (void)ws_size;
    const float* emb        = (const float*)d_in[0];
    const float* ln1_s      = (const float*)d_in[1];
    const float* ln1_b      = (const float*)d_in[2];
    const float* wq         = (const float*)d_in[3];
    const float* bq         = (const float*)d_in[4];
    const float* wk         = (const float*)d_in[5];
    const float* bk         = (const float*)d_in[6];
    const float* wv         = (const float*)d_in[7];
    const float* bv         = (const float*)d_in[8];
    const float* bias_table = (const float*)d_in[9];
    const float* wo         = (const float*)d_in[10];
    const float* bo         = (const float*)d_in[11];
    const float* wp         = (const float*)d_in[12];
    const float* bp         = (const float*)d_in[13];
    const float* ln2_s      = (const float*)d_in[14];
    const float* ln2_b      = (const float*)d_in[15];
    const float* w1         = (const float*)d_in[16];
    const float* b1         = (const float*)d_in[17];
    const float* w2         = (const float*)d_in[18];
    const float* b2         = (const float*)d_in[19];
    const int*   rel_pos    = (const int*)d_in[20];
    const unsigned char* attn_mask = (const unsigned char*)d_in[21];
    float* out = (float*)d_out;

    // workspace layout
    unsigned int* wsp   = (unsigned int*)d_ws;
    float*        cbias = (float*)((char*)d_ws + 393216);  // 4*4*4096 f32
    float*        bproj = (float*)((char*)d_ws + 655360);  // 128 f32
    float*        bqs   = (float*)((char*)d_ws + 655872);  // 128 f32
    const unsigned int* wqP    = wsp;
    const unsigned int* wkP    = wsp + 8192;
    const unsigned int* wvP    = wsp + 16384;
    const unsigned int* wprojP = wsp + 24576;
    const unsigned int* w1P    = wsp + 32768;
    const unsigned int* w2P    = wsp + 65536;

    prep_kernel<<<512, 256, 0, stream>>>(wq, wk, wv, wo, wp, bo, bp, w1, w2,
                                         bias_table, rel_pos, bq, attn_mask,
                                         wsp, cbias, bproj, bqs);

    attn_kernel<<<32 * 7 * 7, 256, 0, stream>>>(emb, ln1_s, ln1_b, bqs, bk, bv,
                                                wqP, wkP, wvP, wprojP, cbias,
                                                bproj, out);

    mlp_kernel<<<(32 * 56 * 56) / 32, 256, 0, stream>>>(out, ln2_s, ln2_b, w1P,
                                                        b1, w2P, b2);
}
